// RBFLayer_44521630990797
// MI455X (gfx1250) — matrix operand order repaired and checked by reading, not compiled, not run on hardware
//
#include <hip/hip_runtime.h>

// RBF layer: out[b,k] = exp(-beta[k] * (||x_b||^2 + ||c_k||^2 - 2 x_b . c_k))
// B=16384, D=256, K=1024.  Cross term done as bf16 WMMA GEMM (f32 accumulate).
// Tile staging uses CDNA5 GLOBAL_LOAD_ASYNC_TO_LDS (ASYNCcnt) when available.

#define RBF_B 16384
#define RBF_D 256
#define RBF_K 1024

typedef __attribute__((ext_vector_type(4)))  float  vfloat4;
typedef __attribute__((ext_vector_type(4)))  int    v4i;
typedef __attribute__((ext_vector_type(8)))  float  v8f;
typedef __attribute__((ext_vector_type(16))) __bf16 v16bf;

union Frag {
    vfloat4 f4[2];
    v16bf   v;
};

#if __has_builtin(__builtin_amdgcn_global_load_async_to_lds_b128)
#define RBF_ASYNC_LDS 1
#else
#define RBF_ASYNC_LDS 0
#endif

#if RBF_ASYNC_LDS
#define RBF_AS1 __attribute__((address_space(1)))
#define RBF_AS3 __attribute__((address_space(3)))
__device__ __forceinline__ void rbf_async_cp16(const unsigned short* g, unsigned short* l) {
    // per-lane: LDS[l] <= MEM[g], 16 bytes, tracked with ASYNCcnt
    __builtin_amdgcn_global_load_async_to_lds_b128((RBF_AS1 v4i*)g, (RBF_AS3 v4i*)l,
                                                   /*offset=*/0, /*cpol=*/0);
}
__device__ __forceinline__ void rbf_wait_async0() {
#if __has_builtin(__builtin_amdgcn_s_wait_asynccnt)
    __builtin_amdgcn_s_wait_asynccnt(0);
#else
    asm volatile("s_wait_asynccnt 0" ::: "memory");
#endif
}
#endif

// ---------------------------------------------------------------------------
// Prepass: per-row squared norm + f32 -> bf16 (round-to-nearest-even) convert.
// One 256-thread block per row (D == 256).
// ---------------------------------------------------------------------------
__global__ __launch_bounds__(256)
void rbf_rowsq_convert(const float* __restrict__ in,
                       unsigned short* __restrict__ out_bf,
                       float* __restrict__ out_sq) {
    const int row = blockIdx.x;
    const int tid = threadIdx.x;

    const float v = in[(size_t)row * RBF_D + tid];

    // f32 -> bf16 RNE
    unsigned int u = __float_as_uint(v);
    unsigned int r = u + 0x7FFFu + ((u >> 16) & 1u);
    out_bf[(size_t)row * RBF_D + tid] = (unsigned short)(r >> 16);

    // block reduction of v*v (wave32 shuffles, then cross-wave via LDS)
    float s = v * v;
    #pragma unroll
    for (int off = 16; off > 0; off >>= 1)
        s += __shfl_down(s, off, 32);

    __shared__ float ws[8];
    if ((tid & 31) == 0) ws[tid >> 5] = s;
    __syncthreads();
    if (tid < 8) {
        float t = ws[tid];
        #pragma unroll
        for (int off = 4; off > 0; off >>= 1)
            t += __shfl_down(t, off, 8);
        if (tid == 0) out_sq[row] = t;
    }
}

// ---------------------------------------------------------------------------
// Main GEMM + RBF epilogue.
// Workgroup: 256 threads = 8 waves, computes a 128(M) x 128(N) output tile.
// Waves arranged 2(M) x 4(N); each wave owns 64x32 = 4x2 WMMA accumulators.
// K-loop: 8 steps of 32 (bf16 16x16x32 WMMA), double-buffered LDS tiles.
// LDS row stride padded to 80B (40 bf16) to spread banks on b128 loads.
// ---------------------------------------------------------------------------
#define LS 40   // bf16 elements per LDS tile row (32 data + 8 pad)

__global__ __launch_bounds__(256)
void rbf_gemm_wmma(const unsigned short* __restrict__ xbf,
                   const unsigned short* __restrict__ cbf,
                   const float* __restrict__ xsq,
                   const float* __restrict__ csq,
                   const float* __restrict__ betas,
                   float* __restrict__ out) {
    __shared__ alignas(16) unsigned short lA[2][128 * LS];
    __shared__ alignas(16) unsigned short lB[2][128 * LS];

    const int tid  = threadIdx.x;
    const int lane = tid & 31;
    const int wave = tid >> 5;
    const int wm   = (wave >> 2) * 64;   // wave M offset in tile (0 or 64)
    const int wn   = (wave & 3) * 32;    // wave N offset in tile (0..96)
    const int bm   = blockIdx.x * 128;   // block M base (x rows)
    const int bn   = blockIdx.y * 128;   // block N base (centers)

    // Global->LDS staging: 128 rows x 64B per tile = 512 x b128; 2 per thread.
    const int gr0  = tid >> 2;           // row 0..63 (second pass +64)
    const int gseg = (tid & 3) * 8;      // element offset within 32-elem row

    const v8f vzero = {};
    v8f acc[4][2];
    #pragma unroll
    for (int i = 0; i < 4; ++i)
        #pragma unroll
        for (int j = 0; j < 2; ++j)
            acc[i][j] = vzero;

    const int am   = lane & 15;          // A: row within 16x32 subtile
    const int akb  = (lane >> 4) * 8;    // A: K chunk base (0 or 8)
    const int bn16 = lane & 15;          // B: column (center) within subtile
    const int bkb  = (lane >> 4) * 8;    // B: K chunk base (0 or 8)

#if RBF_ASYNC_LDS
    // ---------------- async global->LDS path (ASYNCcnt) ----------------
    // Preload k-step 0 into buffer 0
    #pragma unroll
    for (int i = 0; i < 2; ++i) {
        rbf_async_cp16(xbf + (size_t)(bm + gr0 + 64 * i) * RBF_D + gseg,
                       &lA[0][(gr0 + 64 * i) * LS + gseg]);
        rbf_async_cp16(cbf + (size_t)(bn + gr0 + 64 * i) * RBF_D + gseg,
                       &lB[0][(gr0 + 64 * i) * LS + gseg]);
    }
    rbf_wait_async0();
    __syncthreads();

    for (int kt = 0; kt < 8; ++kt) {
        const int buf = kt & 1;

        // Kick off next tile's async copies; they DMA while WMMAs run.
        if (kt < 7) {
            const int kofs = (kt + 1) * 32;
            #pragma unroll
            for (int i = 0; i < 2; ++i) {
                rbf_async_cp16(xbf + (size_t)(bm + gr0 + 64 * i) * RBF_D + kofs + gseg,
                               &lA[buf ^ 1][(gr0 + 64 * i) * LS + gseg]);
                rbf_async_cp16(cbf + (size_t)(bn + gr0 + 64 * i) * RBF_D + kofs + gseg,
                               &lB[buf ^ 1][(gr0 + 64 * i) * LS + gseg]);
            }
        }

        // Load fragments from LDS (layouts per CDNA5 ISA 7.12.2)
        Frag a[4], b[2];
        #pragma unroll
        for (int i = 0; i < 4; ++i) {
            const unsigned short* base = &lA[buf][(wm + i * 16 + am) * LS + akb];
            a[i].f4[0] = *(const vfloat4*)(base);        // K = akb .. akb+7
            a[i].f4[1] = *(const vfloat4*)(base + 16);   // K = akb+16 .. akb+23
        }
        #pragma unroll
        for (int j = 0; j < 2; ++j) {
            const unsigned short* base = &lB[buf][(wn + j * 16 + bn16) * LS + bkb];
            b[j].f4[0] = *(const vfloat4*)(base);        // K = bkb .. bkb+7
            b[j].f4[1] = *(const vfloat4*)(base + 16);   // K = bkb+16 .. bkb+23
        }

        #pragma unroll
        for (int i = 0; i < 4; ++i)
            #pragma unroll
            for (int j = 0; j < 2; ++j)
                acc[i][j] = __builtin_amdgcn_wmma_f32_16x16x32_bf16(
                    false, a[i].v, false, b[j].v,
                    (short)0, acc[i][j], false, false);

        if (kt < 7) rbf_wait_async0();   // own wave's copies into buf^1 done
        __syncthreads();                 // all waves' copies done / reads done
    }
#else
    // ---------------- fallback: VGPR staging path ----------------
    vfloat4 ra[2], rb[2];
    #pragma unroll
    for (int i = 0; i < 2; ++i) {
        ra[i] = *(const vfloat4*)(xbf + (size_t)(bm + gr0 + 64 * i) * RBF_D + gseg);
        rb[i] = *(const vfloat4*)(cbf + (size_t)(bn + gr0 + 64 * i) * RBF_D + gseg);
    }
    #pragma unroll
    for (int i = 0; i < 2; ++i) {
        *(vfloat4*)&lA[0][(gr0 + 64 * i) * LS + gseg] = ra[i];
        *(vfloat4*)&lB[0][(gr0 + 64 * i) * LS + gseg] = rb[i];
    }
    __syncthreads();

    for (int kt = 0; kt < 8; ++kt) {
        const int buf = kt & 1;
        if (kt < 7) {
            const int kofs = (kt + 1) * 32;
            #pragma unroll
            for (int i = 0; i < 2; ++i) {
                ra[i] = *(const vfloat4*)(xbf + (size_t)(bm + gr0 + 64 * i) * RBF_D + kofs + gseg);
                rb[i] = *(const vfloat4*)(cbf + (size_t)(bn + gr0 + 64 * i) * RBF_D + kofs + gseg);
            }
        }

        Frag a[4], b[2];
        #pragma unroll
        for (int i = 0; i < 4; ++i) {
            const unsigned short* base = &lA[buf][(wm + i * 16 + am) * LS + akb];
            a[i].f4[0] = *(const vfloat4*)(base);
            a[i].f4[1] = *(const vfloat4*)(base + 16);
        }
        #pragma unroll
        for (int j = 0; j < 2; ++j) {
            const unsigned short* base = &lB[buf][(wn + j * 16 + bn16) * LS + bkb];
            b[j].f4[0] = *(const vfloat4*)(base);
            b[j].f4[1] = *(const vfloat4*)(base + 16);
        }

        #pragma unroll
        for (int i = 0; i < 4; ++i)
            #pragma unroll
            for (int j = 0; j < 2; ++j)
                acc[i][j] = __builtin_amdgcn_wmma_f32_16x16x32_bf16(
                    false, a[i].v, false, b[j].v,
                    (short)0, acc[i][j], false, false);

        if (kt < 7) {
            #pragma unroll
            for (int i = 0; i < 2; ++i) {
                *(vfloat4*)&lA[buf ^ 1][(gr0 + 64 * i) * LS + gseg] = ra[i];
                *(vfloat4*)&lB[buf ^ 1][(gr0 + 64 * i) * LS + gseg] = rb[i];
            }
        }
        __syncthreads();
    }
#endif

    // Epilogue: C/D layout => col = lane&15, row = 8*(lane>=16) + vgpr
    const int ecol = lane & 15;
    const int erow = (lane >> 4) * 8;
    #pragma unroll
    for (int j = 0; j < 2; ++j) {
        const int col = bn + wn + j * 16 + ecol;
        const float cs = csq[col];
        const float bt = betas[col];
        #pragma unroll
        for (int i = 0; i < 4; ++i) {
            const int rbase = bm + wm + i * 16 + erow;
            #pragma unroll
            for (int v = 0; v < 8; ++v) {
                const int row = rbase + v;
                const float dn = xsq[row] + cs - 2.0f * acc[i][j][v];
                out[(size_t)row * RBF_K + col] = __expf(-bt * dn);
            }
        }
    }
}

// ---------------------------------------------------------------------------
// Launch
// ---------------------------------------------------------------------------
extern "C" void kernel_launch(void* const* d_in, const int* in_sizes, int n_in,
                              void* d_out, int out_size, void* d_ws, size_t ws_size,
                              hipStream_t stream) {
    const float* x       = (const float*)d_in[0];   // [16384, 256]
    const float* centers = (const float*)d_in[1];   // [1024, 256]
    const float* betas   = (const float*)d_in[2];   // [1024]
    float* out = (float*)d_out;                     // [16384, 1024]

    // Workspace: xbf (8 MB) | cbf (0.5 MB) | xsq (64 KB) | csq (4 KB)
    unsigned short* xbf = (unsigned short*)d_ws;
    unsigned short* cbf = xbf + (size_t)RBF_B * RBF_D;
    float* xsq = (float*)(cbf + (size_t)RBF_K * RBF_D);
    float* csq = xsq + RBF_B;

    rbf_rowsq_convert<<<RBF_B, 256, 0, stream>>>(x, xbf, xsq);
    rbf_rowsq_convert<<<RBF_K, 256, 0, stream>>>(centers, cbf, csq);

    dim3 grid(RBF_B / 128, RBF_K / 128);  // 128 x 8
    rbf_gemm_wmma<<<grid, 256, 0, stream>>>(xbf, cbf, xsq, csq, betas, out);
}
